// INR_18597208391727
// MI455X (gfx1250) — matrix operand order repaired and checked by reading, not compiled, not run on hardware
//
#include <hip/hip_runtime.h>

// ---------------------------------------------------------------------------
// Fused Gabor-MLP INR forward for gfx1250 (MI455X).
//   h = cos(10*lin) * exp(-(10*lin)^2),  lin = h_prev @ W.T + b
// Activations LDS-resident; v_wmma_f32_16x16x32_f16 does the GEMMs.
// Per k-step: batch-load A + all 8 B fragments (all live at once) so the
// scheduler overlaps ds_load latency with the matrix pipe (partial dscnt
// waits instead of full drains).
// ---------------------------------------------------------------------------

typedef __attribute__((ext_vector_type(16))) _Float16 v16h;
typedef __attribute__((ext_vector_type(8)))  _Float16 v8h;
typedef __attribute__((ext_vector_type(8)))  float    v8f;

#define HID            256
#define ROWS_PER_BLOCK 64
#define THREADS        256
#define ACT_COLS       288   // 256 hidden + 3 skip(x) + zero pad, multiple of 32
#define ACT_STRIDE     296   // +8 halfs -> 592B row stride, conflict-free
#define W_PAD          8     // weight LDS row pad (halfs)

// f16 weight arena layout in d_ws (element offsets, halfs):
//   W0h [256][ 32]  @ 0        (K=3   padded to 32)
//   W1h [256][256]  @ 8192
//   W2h [256][256]  @ 73728
//   W3h [256][288]  @ 139264   (K=259 padded to 288)
//   W4h [256][256]  @ 212992

__global__ void cvt_weights_f16(const float* __restrict__ W,
                                _Float16* __restrict__ dst,
                                int K, int Kpad) {
  int i = blockIdx.x * blockDim.x + threadIdx.x;
  int total = HID * Kpad;
  if (i >= total) return;
  int o = i / Kpad;
  int k = i - o * Kpad;
  dst[i] = (k < K) ? (_Float16)W[o * K + k] : (_Float16)0.0f;
}

__device__ __forceinline__ v16h cat8(v8h a, v8h b) {
  return __builtin_shufflevector(a, b, 0, 1, 2, 3, 4, 5, 6, 7,
                                       8, 9, 10, 11, 12, 13, 14, 15);
}

// A fragment (16x32 f16): lane r16 = row; lanes 0-15 hold K {kb..+7, kb+16..+23},
// lanes 16-31 hold K {kb+8..+15, kb+24..+31}. Two 16B LDS loads.
__device__ __forceinline__ v16h load_afrag(const _Float16* arow, int kb, int hi) {
  v8h a0 = *(const v8h*)(arow + kb + hi * 8);
  v8h a1 = *(const v8h*)(arow + kb + 16 + hi * 8);
  return cat8(a0, a1);
}

// B fragment (32x16 f16): lane r16 = output neuron; K {kb+hi*8..+7, kb+16+hi*8..+7}.
__device__ __forceinline__ v16h load_bfrag(const _Float16* wlds, int n,
                                           int KS, int kb, int hi, int r16) {
  const _Float16* brow = wlds + (n + r16) * KS + kb + hi * 8;
  v8h b0 = *(const v8h*)(brow);
  v8h b1 = *(const v8h*)(brow + 16);
  return cat8(b0, b1);
}

__global__ __launch_bounds__(THREADS)
void gabor_mlp_fused(const float* __restrict__ x,
                     const _Float16* __restrict__ Wh,
                     const float* __restrict__ b0, const float* __restrict__ b1,
                     const float* __restrict__ b2, const float* __restrict__ b3,
                     const float* __restrict__ b4,
                     const float* __restrict__ Wf, const float* __restrict__ bf,
                     float* __restrict__ out) {
  extern __shared__ _Float16 smem[];
  _Float16* act  = smem;                                      // [64][ACT_STRIDE]
  _Float16* wlds = smem + ROWS_PER_BLOCK * ACT_STRIDE;        // [256][Kpad+W_PAD]
  float*    blds = (float*)(wlds + HID * (ACT_COLS + W_PAD)); // [256]

  const int tid   = threadIdx.x;
  const int lane  = tid & 31;
  const int wave  = tid >> 5;
  const int r16   = lane & 15;      // row-in-tile / col-in-tile
  const int hi    = lane >> 4;      // K-half selector (ISA A/B lane layout)
  const int strip = wave >> 1;      // which 16-row strip (0..3)
  const int tb    = (wave & 1) * 8; // which 8 output tiles (0..7 or 8..15)

  // ---- init: zero act, stash x into cols 0..2 and 256..258 -----------------
  {
    uint4 z = {0u, 0u, 0u, 0u};
    uint4* av = (uint4*)act;
    const int nvec = ROWS_PER_BLOCK * ACT_STRIDE / 8;
    for (int i = tid; i < nvec; i += THREADS) av[i] = z;
  }
  __syncthreads();
  if (tid < ROWS_PER_BLOCK) {
    long g = (long)blockIdx.x * ROWS_PER_BLOCK + tid;
    float x0 = x[g * 3 + 0], x1 = x[g * 3 + 1], x2 = x[g * 3 + 2];
    _Float16* r = act + tid * ACT_STRIDE;
    r[0]   = (_Float16)x0; r[1]   = (_Float16)x1; r[2]   = (_Float16)x2;
    r[256] = (_Float16)x0; r[257] = (_Float16)x1; r[258] = (_Float16)x2;
  }
  __syncthreads();

  // ---- five Gabor layers ---------------------------------------------------
  long woff = 0;
  for (int L = 0; L < 5; ++L) {
    const int Kpad = (L == 0) ? 32 : ((L == 3) ? 288 : 256);
    const int KS   = Kpad + W_PAD;
    const float* bp = (L == 0) ? b0 : (L == 1) ? b1 : (L == 2) ? b2
                                    : (L == 3) ? b3 : b4;

    // stage f16 weights [out][Kpad] into padded LDS, bias into LDS
    {
      const int rowvec = Kpad >> 3;  // uint4s per source row
      const uint4* src = (const uint4*)(Wh + woff);
      for (int i = tid; i < HID * rowvec; i += THREADS) {
        int row = i / rowvec;
        int col = i - row * rowvec;
        *(uint4*)(wlds + row * KS + col * 8) = src[i];
      }
      blds[tid] = bp[tid];
    }
    __syncthreads();

    // 16 rows (this wave's strip) x 128 outputs (this wave's half)
    v8f acc[8];
#pragma unroll
    for (int t = 0; t < 8; ++t) acc[t] = (v8f){0, 0, 0, 0, 0, 0, 0, 0};

    const _Float16* arow = act + (strip * 16 + r16) * ACT_STRIDE;
    const int ksteps = Kpad >> 5;

    for (int ks = 0; ks < ksteps; ++ks) {
      const int kb = ks * 32;
      // Batch-load: A + all 8 B fragments live simultaneously (18 ds_load_b128)
      v16h a = load_afrag(arow, kb, hi);
      v16h bf0 = load_bfrag(wlds, (tb + 0) * 16, KS, kb, hi, r16);
      v16h bf1 = load_bfrag(wlds, (tb + 1) * 16, KS, kb, hi, r16);
      v16h bf2 = load_bfrag(wlds, (tb + 2) * 16, KS, kb, hi, r16);
      v16h bf3 = load_bfrag(wlds, (tb + 3) * 16, KS, kb, hi, r16);
      v16h bf4 = load_bfrag(wlds, (tb + 4) * 16, KS, kb, hi, r16);
      v16h bf5 = load_bfrag(wlds, (tb + 5) * 16, KS, kb, hi, r16);
      v16h bf6 = load_bfrag(wlds, (tb + 6) * 16, KS, kb, hi, r16);
      v16h bf7 = load_bfrag(wlds, (tb + 7) * 16, KS, kb, hi, r16);
      acc[0] = __builtin_amdgcn_wmma_f32_16x16x32_f16(false, a, false, bf0,
                                                      (short)0, acc[0], false, false);
      acc[1] = __builtin_amdgcn_wmma_f32_16x16x32_f16(false, a, false, bf1,
                                                      (short)0, acc[1], false, false);
      acc[2] = __builtin_amdgcn_wmma_f32_16x16x32_f16(false, a, false, bf2,
                                                      (short)0, acc[2], false, false);
      acc[3] = __builtin_amdgcn_wmma_f32_16x16x32_f16(false, a, false, bf3,
                                                      (short)0, acc[3], false, false);
      acc[4] = __builtin_amdgcn_wmma_f32_16x16x32_f16(false, a, false, bf4,
                                                      (short)0, acc[4], false, false);
      acc[5] = __builtin_amdgcn_wmma_f32_16x16x32_f16(false, a, false, bf5,
                                                      (short)0, acc[5], false, false);
      acc[6] = __builtin_amdgcn_wmma_f32_16x16x32_f16(false, a, false, bf6,
                                                      (short)0, acc[6], false, false);
      acc[7] = __builtin_amdgcn_wmma_f32_16x16x32_f16(false, a, false, bf7,
                                                      (short)0, acc[7], false, false);
    }
    __syncthreads();   // all waves done READING act before overwrite

    // Gabor activation in registers, write back f16 to act cols 0..255
#pragma unroll
    for (int t = 0; t < 8; ++t) {
      float bc = blds[(tb + t) * 16 + r16];
#pragma unroll
      for (int j = 0; j < 8; ++j) {
        int row = strip * 16 + hi * 8 + j;   // C/D layout: VGPR j -> M=j(+8)
        float s = 10.0f * (acc[t][j] + bc);  // OMEGA == SIGMA == 10
        float g = __cosf(s) * __expf(-s * s);
        act[row * ACT_STRIDE + (tb + t) * 16 + r16] = (_Float16)g;
      }
    }
    __syncthreads();

    woff += (long)HID * Kpad;
  }

  // ---- final linear 256 -> 3 (negligible FLOPs, VALU) ----------------------
  if (tid < ROWS_PER_BLOCK) {
    const _Float16* r = act + tid * ACT_STRIDE;
    float s0 = bf[0], s1 = bf[1], s2 = bf[2];
    for (int k = 0; k < HID; ++k) {
      float a = (float)r[k];
      s0 += a * Wf[k];
      s1 += a * Wf[HID + k];
      s2 += a * Wf[2 * HID + k];
    }
    long g = (long)blockIdx.x * ROWS_PER_BLOCK + tid;
    out[g * 3 + 0] = s0;
    out[g * 3 + 1] = s1;
    out[g * 3 + 2] = s2;
  }
}

extern "C" void kernel_launch(void* const* d_in, const int* in_sizes, int n_in,
                              void* d_out, int out_size, void* d_ws, size_t ws_size,
                              hipStream_t stream) {
  const float* x  = (const float*)d_in[0];
  const float* W0 = (const float*)d_in[1];
  const float* b0 = (const float*)d_in[2];
  const float* W1 = (const float*)d_in[3];
  const float* b1 = (const float*)d_in[4];
  const float* W2 = (const float*)d_in[5];
  const float* b2 = (const float*)d_in[6];
  const float* W3 = (const float*)d_in[7];
  const float* b3 = (const float*)d_in[8];
  const float* W4 = (const float*)d_in[9];
  const float* b4 = (const float*)d_in[10];
  const float* Wf = (const float*)d_in[11];
  const float* bf = (const float*)d_in[12];

  _Float16* Wh = (_Float16*)d_ws;
  const int rows = in_sizes[0] / 3;           // 262144

  auto cvt = [&](const float* W, long off, int K, int Kpad) {
    int total = HID * Kpad;
    cvt_weights_f16<<<(total + THREADS - 1) / THREADS, THREADS, 0, stream>>>(
        W, Wh + off, K, Kpad);
  };
  cvt(W0, 0,      3,   32);
  cvt(W1, 8192,   256, 256);
  cvt(W2, 73728,  256, 256);
  cvt(W3, 139264, 259, 288);
  cvt(W4, 212992, 256, 256);

  size_t smem = (size_t)(ROWS_PER_BLOCK * ACT_STRIDE + HID * (ACT_COLS + W_PAD)) *
                    sizeof(_Float16) +
                HID * sizeof(float);
  (void)hipFuncSetAttribute((const void*)gabor_mlp_fused,
                            hipFuncAttributeMaxDynamicSharedMemorySize,
                            (int)smem);

  gabor_mlp_fused<<<rows / ROWS_PER_BLOCK, THREADS, smem, stream>>>(
      x, Wh, b0, b1, b2, b3, b4, Wf, bf, (float*)d_out);
}
